// AstrometryConcordanceHead_90159953477958
// MI455X (gfx1250) — matrix operand order repaired and checked by reading, not compiled, not run on hardware
//
#include <hip/hip_runtime.h>
#include <hip/hip_bf16.h>

typedef __bf16 v16bf __attribute__((ext_vector_type(16)));
typedef __bf16 v8bf  __attribute__((ext_vector_type(8)));
typedef float  v8f   __attribute__((ext_vector_type(8)));

#define RR   3
#define KOFF 49
#define Bn   8
#define Dd   256
#define Hh   192
#define Ww   192
#define Pp   (Hh * Ww)   /* 36864 */
#define Dm   64
#define WPAD 264         /* padded bf16 row stride for LDS tiles */

static __device__ __forceinline__ unsigned short f2bf(float x) {
    unsigned int u = __float_as_uint(x);
    unsigned int r = u + 0x7fffu + ((u >> 16) & 1u);   // round-to-nearest-even
    return (unsigned short)(r >> 16);
}
#define BLO(u) __uint_as_float((u) << 16)
#define BHI(u) __uint_as_float((u) & 0xffff0000u)

// ---------------------------------------------------------------------------
// Kernel 1: fused projection GEMM (WMMA bf16) + L2 normalize.
// Block = 256 threads (8 waves), owns 64 consecutive pixels of one batch.
// C[64m x 64p] = W[64 x 256] * X[256 x 64], done as 4x4 tiles of
// v_wmma_f32_16x16x32_bf16 over 8 K-steps.
// ---------------------------------------------------------------------------
__global__ void proj_norm_kernel(const float* __restrict__ rub_in,
                                 const float* __restrict__ vis_in,
                                 const float* __restrict__ Wr,
                                 const float* __restrict__ Wv,
                                 unsigned short* __restrict__ outR,
                                 unsigned short* __restrict__ outV) {
    extern __shared__ __align__(16) char smem[];
    unsigned short* sW = (unsigned short*)smem;          // [2][64][WPAD] bf16
    unsigned short* sX = sW + 2 * Dm * WPAD;             // [64 p][WPAD] bf16
    float* sC    = (float*)(sX + 64 * WPAD);             // [64 m][65] f32
    float* sNorm = sC + 64 * 65;                         // [64 p]

    const int t    = threadIdx.x;
    const int lane = t & 31;
    const int wv   = t >> 5;
    const int blk  = blockIdx.x;
    const int b    = blk / (Pp / 64);
    const int p0   = (blk % (Pp / 64)) * 64;

    // Load both projection matrices as bf16 [m][d] (hits L2 after first use).
    for (int idx = t; idx < Dm * Dd; idx += 256) {
        int m = idx >> 8, d = idx & 255;
        sW[0 * Dm * WPAD + m * WPAD + d] = f2bf(Wr[idx]);
        sW[1 * Dm * WPAD + m * WPAD + d] = f2bf(Wv[idx]);
    }

    for (int inp = 0; inp < 2; ++inp) {
        __syncthreads();
        const float* Xg = (inp == 0 ? rub_in : vis_in) + (size_t)b * Dd * Pp + p0;
        // Coalesced load of the 256x64 f32 tile, transposed into LDS as [p][d] bf16.
        for (int idx = t; idx < Dd * 64; idx += 256) {
            int d = idx >> 6, p = idx & 63;
            if (d + 8 < Dd) __builtin_prefetch(Xg + (size_t)(d + 8) * Pp + p, 0, 1);
            sX[p * WPAD + d] = f2bf(Xg[(size_t)d * Pp + p]);
        }
        __syncthreads();

        const unsigned short* sWi = sW + inp * Dm * WPAD;
        // 16 output tiles (mt,nt), 2 per wave.
        for (int tt = 2 * wv; tt < 2 * wv + 2; ++tt) {
            const int mt = tt >> 2, nt = tt & 3;
            v8f acc = {};
            const int lo = (lane < 16) ? 0 : 8;               // A K-offset half
            const int m  = mt * 16 + (lane & 15);             // A row
            const int pl = nt * 16 + (lane & 15);             // B column (pixel)
            const int kboff = (lane < 16) ? 0 : 8;            // B K-offset half
#pragma unroll
            for (int ks = 0; ks < 8; ++ks) {
                union { v16bf v; v8bf h[2]; } A, Bf;
                const unsigned short* ap = sWi + m * WPAD + ks * 32 + lo;
                A.h[0] = *(const v8bf*)ap;            // K = base .. base+7
                A.h[1] = *(const v8bf*)(ap + 16);     // K = base+16 .. base+23
                const unsigned short* bp = sX + pl * WPAD + ks * 32 + kboff;
                Bf.h[0] = *(const v8bf*)bp;           // K = base .. base+7
                Bf.h[1] = *(const v8bf*)(bp + 16);    // K = base+16 .. base+23
                acc = __builtin_amdgcn_wmma_f32_16x16x32_bf16(
                    false, A.v, false, Bf.v, (short)0, acc, false, false);
            }
            // Scatter C fragment: lane n = lane&15, rows v + (lane<16 ? 0 : 8).
            const int n = lane & 15;
            const int mrow0 = mt * 16 + ((lane < 16) ? 0 : 8);
#pragma unroll
            for (int v = 0; v < 8; ++v)
                sC[(mrow0 + v) * 65 + nt * 16 + n] = acc[v];
        }
        __syncthreads();

        if (t < 64) {  // per-pixel L2 norm over 64 channels
            float s = 0.f;
#pragma unroll 8
            for (int m = 0; m < 64; ++m) { float x = sC[m * 65 + t]; s += x * x; }
            sNorm[t] = 1.0f / fmaxf(sqrtf(s), 1e-6f);
        }
        __syncthreads();

        unsigned short* outp = (inp == 0 ? outR : outV);
        for (int idx = t; idx < 64 * 64; idx += 256) {   // m fastest -> coalesced
            int p = idx >> 6, m = idx & 63;
            float val = sC[m * 65 + p] * sNorm[p];
            outp[((size_t)(b * Pp + p0 + p)) * 64 + m] = f2bf(val);
        }
    }
}

// ---------------------------------------------------------------------------
// Kernel 2: 7x7 correlation + local softmax + per-(b,k) logit accumulation.
// Block = 256 threads = one 16x16 pixel tile; vis halo (22x22x64 bf16) in LDS.
// ---------------------------------------------------------------------------
static __device__ __forceinline__ float dot64(const float* __restrict__ r,
                                              const unsigned short* vptr) {
    const uint4* vp = (const uint4*)vptr;
    float acc = 0.f;
#pragma unroll
    for (int j = 0; j < 8; ++j) {
        uint4 u = vp[j];
        acc += r[j * 8 + 0] * BLO(u.x) + r[j * 8 + 1] * BHI(u.x);
        acc += r[j * 8 + 2] * BLO(u.y) + r[j * 8 + 3] * BHI(u.y);
        acc += r[j * 8 + 4] * BLO(u.z) + r[j * 8 + 5] * BHI(u.z);
        acc += r[j * 8 + 6] * BLO(u.w) + r[j * 8 + 7] * BHI(u.w);
    }
    return acc;
}

__global__ void corr_softmax_kernel(const unsigned short* __restrict__ rubN,
                                    const unsigned short* __restrict__ visN,
                                    const float* __restrict__ logtemp,
                                    float* __restrict__ out,
                                    float* __restrict__ gsum) {
    __shared__ unsigned short sVis[22 * 22 * 64];
    __shared__ float sRed[KOFF];
    const int t  = threadIdx.x;
    const int bx = blockIdx.x % (Ww / 16);
    const int by = (blockIdx.x / (Ww / 16)) % (Hh / 16);
    const int b  = blockIdx.x / ((Ww / 16) * (Hh / 16));
    const int y0 = by * 16, x0 = bx * 16;

    // Halo load with edge clamping.
    for (int idx = t; idx < 22 * 22; idx += 256) {
        int hy = idx / 22, hx = idx % 22;
        int gy = min(max(y0 + hy - RR, 0), Hh - 1);
        int gx = min(max(x0 + hx - RR, 0), Ww - 1);
        const uint4* src = (const uint4*)(visN + ((size_t)(b * Pp + gy * Ww + gx)) * 64);
        uint4* dst = (uint4*)(sVis + idx * 64);
#pragma unroll
        for (int j = 0; j < 8; ++j) dst[j] = src[j];
    }
    if (t < KOFF) sRed[t] = 0.f;
    __syncthreads();

    const int ty = t >> 4, tx = t & 15;
    const int gy = y0 + ty, gx = x0 + tx;

    float r[64];  // own rub vector, bf16 -> f32 via bit shift (exact)
    {
        const uint4* src = (const uint4*)(rubN + ((size_t)(b * Pp + gy * Ww + gx)) * 64);
#pragma unroll
        for (int j = 0; j < 8; ++j) {
            uint4 u = src[j];
            r[j * 8 + 0] = BLO(u.x); r[j * 8 + 1] = BHI(u.x);
            r[j * 8 + 2] = BLO(u.y); r[j * 8 + 3] = BHI(u.y);
            r[j * 8 + 4] = BLO(u.z); r[j * 8 + 5] = BHI(u.z);
            r[j * 8 + 6] = BLO(u.w); r[j * 8 + 7] = BHI(u.w);
        }
    }

    const float tau = fmaxf(__expf(logtemp[0]), 1e-3f);

    // Pass 1: max logit + accumulate raw logits into block-level sums.
    float mx = -1e30f;
    for (int dy = -RR; dy <= RR; ++dy)
        for (int dx = -RR; dx <= RR; ++dx) {
            float l = 0.125f * dot64(r, sVis + (((ty + RR + dy) * 22) + (tx + RR + dx)) * 64);
            mx = fmaxf(mx, l);
            atomicAdd(&sRed[(dy + RR) * 7 + (dx + RR)], l);
        }
    // Pass 2: recompute logits, softmax moments (cheap: compute-bound slack is huge).
    float s = 0.f, sdy = 0.f, sdx = 0.f;
    for (int dy = -RR; dy <= RR; ++dy)
        for (int dx = -RR; dx <= RR; ++dx) {
            float l = 0.125f * dot64(r, sVis + (((ty + RR + dy) * 22) + (tx + RR + dx)) * 64);
            float e = __expf((l - mx) / tau);
            s += e; sdy += e * (float)dy; sdx += e * (float)dx;
        }
    float inv  = 1.f / s;
    float conf = inv;            // max prob = exp(0)/s
    float dyl = sdy * inv, dxl = sdx * inv;
    float u = 1.f / (float)KOFF;
    float lw = fminf(fmaxf((conf - u) / (1.f - u), 0.f), 1.f);

    size_t base = (size_t)b * 5 * Pp + (size_t)gy * Ww + gx;
    out[base + 0 * (size_t)Pp] = dyl;   // blended in kernel 4
    out[base + 1 * (size_t)Pp] = dxl;
    out[base + 2 * (size_t)Pp] = conf;
    out[base + 3 * (size_t)Pp] = lw;

    __syncthreads();
    if (t < KOFF) atomicAdd(&gsum[b * KOFF + t], sRed[t]);
}

// ---------------------------------------------------------------------------
// Kernel 3: global softmax per batch (tiny).
// ---------------------------------------------------------------------------
__global__ void global_softmax_kernel(const float* __restrict__ gsum,
                                      const float* __restrict__ logtemp,
                                      float* __restrict__ gres) {
    __shared__ float l[KOFF];
    const int b = blockIdx.x, t = threadIdx.x;
    if (t < KOFF) l[t] = gsum[b * KOFF + t] * (1.0f / (float)Pp);
    __syncthreads();
    if (t == 0) {
        float tau = fmaxf(__expf(logtemp[0]), 1e-3f);
        float mx = -1e30f;
        for (int k = 0; k < KOFF; ++k) mx = fmaxf(mx, l[k]);
        float s = 0.f, sdy = 0.f, sdx = 0.f;
        for (int k = 0; k < KOFF; ++k) {
            float e = __expf((l[k] - mx) / tau);
            s += e; sdy += e * (float)(k / 7 - RR); sdx += e * (float)(k % 7 - RR);
        }
        gres[b * 3 + 0] = sdy / s;
        gres[b * 3 + 1] = sdx / s;
        gres[b * 3 + 2] = 1.f / s;
    }
}

// ---------------------------------------------------------------------------
// Kernel 4: blend local/global + write conf_global channel.
// ---------------------------------------------------------------------------
__global__ void blend_kernel(float* __restrict__ out, const float* __restrict__ gres) {
    int idx = blockIdx.x * 256 + threadIdx.x;
    if (idx >= Bn * Pp) return;
    int b = idx / Pp, p = idx % Pp;
    size_t base = (size_t)b * 5 * Pp + p;
    float dyl = out[base + 0 * (size_t)Pp];
    float dxl = out[base + 1 * (size_t)Pp];
    float lw  = out[base + 3 * (size_t)Pp];
    float dyg = gres[b * 3 + 0], dxg = gres[b * 3 + 1], cg = gres[b * 3 + 2];
    out[base + 0 * (size_t)Pp] = lw * dyl + (1.f - lw) * dyg;
    out[base + 1 * (size_t)Pp] = lw * dxl + (1.f - lw) * dxg;
    out[base + 4 * (size_t)Pp] = cg;
}

// ---------------------------------------------------------------------------
extern "C" void kernel_launch(void* const* d_in, const int* in_sizes, int n_in,
                              void* d_out, int out_size, void* d_ws, size_t ws_size,
                              hipStream_t stream) {
    const float* rubin = (const float*)d_in[0];
    const float* vis   = (const float*)d_in[1];
    const float* Wr    = (const float*)d_in[2];
    const float* Wv    = (const float*)d_in[3];
    const float* ltmp  = (const float*)d_in[4];
    float* out = (float*)d_out;

    unsigned short* wsR = (unsigned short*)d_ws;
    unsigned short* wsV = wsR + (size_t)Bn * Pp * Dm;
    float* gsum = (float*)(wsV + (size_t)Bn * Pp * Dm);
    float* gres = gsum + Bn * KOFF;

    hipMemsetAsync(gsum, 0, (Bn * KOFF + Bn * 3) * sizeof(float), stream);

    const size_t SM1 = (size_t)(2 * Dm * WPAD + 64 * WPAD) * 2 +
                       (size_t)(64 * 65 + 64) * 4;   // 118272 bytes
    proj_norm_kernel<<<Bn * (Pp / 64), 256, SM1, stream>>>(rubin, vis, Wr, Wv, wsR, wsV);

    corr_softmax_kernel<<<Bn * (Hh / 16) * (Ww / 16), 256, 0, stream>>>(
        wsR, wsV, ltmp, out, gsum);

    global_softmax_kernel<<<Bn, 64, 0, stream>>>(gsum, ltmp, gres);

    blend_kernel<<<(Bn * Pp + 255) / 256, 256, 0, stream>>>(out, gres);
}
